// Net_16767552323923
// MI455X (gfx1250) — compile-verified, matrix operand order checked by reading, not run on hardware
//
#include <hip/hip_runtime.h>
#include <hip/hip_bf16.h>

#define B_SZ 2048
#define T_SZ 100
#define I_SZ 784
#define H_SZ 100
#define O_SZ 10
#define IPAD 800
#define HPAD 128
#define OPAD 16
#define NQ   (I_SZ / 4)                 // 196 float4 chunks per row

typedef __bf16 bf16;
typedef __attribute__((ext_vector_type(16))) __bf16 v16bf;
typedef __attribute__((ext_vector_type(8)))  __bf16 bf16x8;
typedef __attribute__((ext_vector_type(4)))  __bf16 bf16x4;
typedef __attribute__((ext_vector_type(8)))  float  v8f;

// LDS layout (bytes); all offsets 16B-aligned
#define W1S_OFF   0
#define W1S_BYTES (HPAD * IPAD * 2)          // 204800
#define XF_OFF    (W1S_OFF + W1S_BYTES)      // fp32 async staging tile [16][IPAD]
#define XF_BYTES  (16 * IPAD * 4)            // 51200
#define XS_OFF    (XF_OFF + XF_BYTES)        // bf16 compute tile [16][IPAD]
#define XS_BYTES  (16 * IPAD * 2)            // 25600
#define SPK_OFF   (XS_OFF + XS_BYTES)        // bf16 spikes [16][HPAD]
#define SPK_BYTES (16 * HPAD * 2)            // 4096
#define W2S_OFF   (SPK_OFF + SPK_BYTES)      // bf16 W2 [OPAD][HPAD]
#define W2S_BYTES (OPAD * HPAD * 2)          // 4096
#define SMEM_BYTES (W2S_OFF + W2S_BYTES)     // 289792

// Load one 16-bit WMMA operand fragment (16 bf16) from an LDS row:
// two ds_load_b128 at +0 and +32 bytes per the ISA 16-bit A/B lane layout.
static __device__ __forceinline__ v16bf load_frag(const bf16* p) {
    bf16x8 lo = *(const bf16x8*)p;
    bf16x8 hi = *(const bf16x8*)(p + 16);
    return __builtin_shufflevector(lo, hi, 0, 1, 2, 3, 4, 5, 6, 7,
                                   8, 9, 10, 11, 12, 13, 14, 15);
}

// Issue async HBM->LDS copy of x[b0:b0+16, t, :] into the xf staging buffer.
// 16 bytes/lane, GVS addressing: saddr = x base, vaddr = u32 byte offset.
static __device__ __forceinline__ void issue_x_async(const float* __restrict__ x,
                                                     int b0, int t, int tid) {
    for (int idx = tid; idx < 16 * NQ; idx += 256) {
        int m = idx / NQ, q = idx - m * NQ;
        unsigned lds_off = (unsigned)(XF_OFF + (m * IPAD + q * 4) * 4);
        unsigned goff = (unsigned)(((size_t)(b0 + m) * T_SZ * I_SZ +
                                    (size_t)t * I_SZ + (size_t)q * 4) * 4);
        asm volatile("global_load_async_to_lds_b128 %0, %1, %2"
                     :: "v"(lds_off), "v"(goff), "s"(x) : "memory");
        // pull t+2 into L2 while t streams into LDS
        if (t + 2 < T_SZ)
            __builtin_prefetch((const char*)x + goff + (size_t)2 * I_SZ * 4, 0, 1);
    }
}

__global__ __launch_bounds__(256) void snn_lif2_kernel(
    const float* __restrict__ x,  const float* __restrict__ W1, const float* __restrict__ b1,
    const float* __restrict__ W2, const float* __restrict__ b2,
    const float* __restrict__ alpha1, const float* __restrict__ beta1,
    const float* __restrict__ alpha2, const float* __restrict__ beta2,
    float* __restrict__ out_spk, float* __restrict__ out_mem2)
{
    extern __shared__ char smem[];
    bf16*  w1s = (bf16*)(smem + W1S_OFF);   // [HPAD][IPAD]
    float* xf  = (float*)(smem + XF_OFF);   // [16][IPAD] fp32 staging
    bf16*  xs  = (bf16*)(smem + XS_OFF);    // [16][IPAD]
    bf16*  sps = (bf16*)(smem + SPK_OFF);   // [16][HPAD]
    bf16*  w2s = (bf16*)(smem + W2S_OFF);   // [OPAD][HPAD]

    const int tid  = threadIdx.x;
    const int wave = tid >> 5;
    const int lane = tid & 31;
    const int half = lane >> 4;        // 0: lanes 0-15, 1: lanes 16-31
    const int mn   = lane & 15;
    const int koff = half * 8;         // K phase per ISA 16-bit A/B layout
    const int b0   = blockIdx.x * 16;  // batch-row base of this workgroup

    // ---- kick off async stream of x(t=0) immediately; stage weights meanwhile ----
    issue_x_async(x, b0, 0, tid);

    for (int idx = tid; idx < HPAD * IPAD; idx += 256) {
        int n = idx / IPAD, k = idx - n * IPAD;
        float v = (n < H_SZ && k < I_SZ) ? W1[n * I_SZ + k] : 0.0f;
        w1s[idx] = (bf16)v;
    }
    for (int idx = tid; idx < OPAD * HPAD; idx += 256) {
        int n = idx / HPAD, k = idx - n * HPAD;
        float v = (n < O_SZ && k < H_SZ) ? W2[n * H_SZ + k] : 0.0f;
        w2s[idx] = (bf16)v;
    }
    {   // zero the K-pad columns (784..799) of xs once; never rewritten
        int m = tid >> 4, k = I_SZ + (tid & 15);
        xs[m * IPAD + k] = (bf16)0.0f;
    }

    const float a1  = fminf(fmaxf(alpha1[0], 0.0f), 1.0f);
    const float be1 = fminf(fmaxf(beta1[0],  0.0f), 1.0f);
    const float a2  = fminf(fmaxf(alpha2[0], 0.0f), 1.0f);
    const float be2 = fminf(fmaxf(beta2[0],  0.0f), 1.0f);

    const int   ntile = wave;                 // wave w owns H columns [16w, 16w+16)
    const int   nglob = ntile * 16 + mn;
    const float b1v   = (nglob < H_SZ) ? b1[nglob] : 0.0f;
    const float b2v   = (mn < O_SZ)    ? b2[mn]    : 0.0f;

    // persistent recurrent state in registers (C layout: VGPR j -> row j + 8*half)
    float syn1[8], mem1[8], syn2[8], mem2[8];
    #pragma unroll
    for (int j = 0; j < 8; ++j) { syn1[j] = 0.f; mem1[j] = 0.f; syn2[j] = 0.f; mem2[j] = 0.f; }

    // drain t=0 stage and convert fp32 -> bf16 compute tile
    asm volatile("s_wait_asynccnt 0x0" ::: "memory");
    __syncthreads();
    for (int idx = tid; idx < 16 * NQ; idx += 256) {
        int m = idx / NQ, q = idx - m * NQ;
        float4 v = *(const float4*)(xf + m * IPAD + q * 4);
        bf16x4 p = { (bf16)v.x, (bf16)v.y, (bf16)v.z, (bf16)v.w };
        *(bf16x4*)(xs + m * IPAD + q * 4) = p;
    }
    __syncthreads();

    const bf16* arow = xs + mn * IPAD + koff;
    const bf16* brow = w1s + (size_t)(ntile * 16 + mn) * IPAD + koff;

    for (int t = 0; t < T_SZ; ++t) {
        // ---- overlap: async-stream x(t+1) into xf while computing t ----
        if (t + 1 < T_SZ) issue_x_async(x, b0, t + 1, tid);

        // ---- layer 1: software-pipelined 25x v_wmma_f32_16x16x32_bf16 ----
        v8f c = {};
        v16bf a_cur = load_frag(arow);
        v16bf b_cur = load_frag(brow);
        #pragma unroll 4
        for (int k = 0; k < IPAD / 32 - 1; ++k) {
            v16bf a_nxt = load_frag(arow + (k + 1) * 32);
            v16bf b_nxt = load_frag(brow + (k + 1) * 32);
            c = __builtin_amdgcn_wmma_f32_16x16x32_bf16(false, a_cur, false, b_cur,
                                                        (short)0, c, false, false);
            a_cur = a_nxt; b_cur = b_nxt;
        }
        c = __builtin_amdgcn_wmma_f32_16x16x32_bf16(false, a_cur, false, b_cur,
                                                    (short)0, c, false, false);

        // ---- LIF layer 1 (reset from previous mem, subtract-reset, thresh=1) ----
        #pragma unroll
        for (int j = 0; j < 8; ++j) {
            int   m     = j + half * 8;
            float cur   = c[j] + b1v;
            float syn_n = a1 * syn1[j] + cur;
            float reset = (mem1[j] > 1.0f) ? 1.0f : 0.0f;   // from previous mem1
            float mem_n = be1 * mem1[j] + syn_n - reset;
            syn1[j] = syn_n; mem1[j] = mem_n;
            float spk = (mem_n > 1.0f) ? 1.0f : 0.0f;
            sps[m * HPAD + nglob] = (bf16)spk;              // exact in bf16
            if (nglob < H_SZ)
                out_spk[(size_t)t * B_SZ * H_SZ + (size_t)(b0 + m) * H_SZ + nglob] = spk;
        }
        __syncthreads();

        // ---- layer 2 on wave 0: spk_tile @ W2^T (4 WMMAs), then leaky state ----
        if (wave == 0) {
            v8f c2 = {};
            #pragma unroll
            for (int k = 0; k < HPAD / 32; ++k) {
                v16bf av = load_frag(sps + mn * HPAD + k * 32 + koff);
                v16bf bv = load_frag(w2s + mn * HPAD + k * 32 + koff);
                c2 = __builtin_amdgcn_wmma_f32_16x16x32_bf16(false, av, false, bv,
                                                             (short)0, c2, false, false);
            }
            #pragma unroll
            for (int j = 0; j < 8; ++j) {
                int   m   = j + half * 8;
                float cur = c2[j] + b2v;
                syn2[j] = a2 * syn2[j] + cur;
                mem2[j] = be2 * mem2[j] + syn2[j];          // reset_mechanism='none'
                if (mn < O_SZ)
                    out_mem2[(size_t)t * B_SZ * O_SZ + (size_t)(b0 + m) * O_SZ + mn] = mem2[j];
            }
        }
        __syncthreads();

        // ---- drain async x(t+1), convert to bf16 compute tile ----
        if (t + 1 < T_SZ) {
            asm volatile("s_wait_asynccnt 0x0" ::: "memory");
            __syncthreads();
            for (int idx = tid; idx < 16 * NQ; idx += 256) {
                int m = idx / NQ, q = idx - m * NQ;
                float4 v = *(const float4*)(xf + m * IPAD + q * 4);
                bf16x4 p = { (bf16)v.x, (bf16)v.y, (bf16)v.z, (bf16)v.w };
                *(bf16x4*)(xs + m * IPAD + q * 4) = p;
            }
            __syncthreads();
        }
    }
}

extern "C" void kernel_launch(void* const* d_in, const int* in_sizes, int n_in,
                              void* d_out, int out_size, void* d_ws, size_t ws_size,
                              hipStream_t stream) {
    (void)in_sizes; (void)n_in; (void)d_ws; (void)ws_size; (void)out_size;
    const float* x   = (const float*)d_in[0];
    const float* W1  = (const float*)d_in[1];
    const float* b1  = (const float*)d_in[2];
    const float* W2  = (const float*)d_in[3];
    const float* b2  = (const float*)d_in[4];
    const float* a1  = (const float*)d_in[5];
    const float* be1 = (const float*)d_in[6];
    const float* a2  = (const float*)d_in[7];
    const float* be2 = (const float*)d_in[8];

    float* out_spk  = (float*)d_out;                                   // [T,B,H]
    float* out_mem2 = out_spk + (size_t)T_SZ * B_SZ * H_SZ;            // [T,B,O]

    hipFuncSetAttribute((const void*)snn_lif2_kernel,
                        hipFuncAttributeMaxDynamicSharedMemorySize, SMEM_BYTES);

    dim3 grid(B_SZ / 16), block(256);
    snn_lif2_kernel<<<grid, block, SMEM_BYTES, stream>>>(
        x, W1, b1, W2, b2, a1, be1, a2, be2, out_spk, out_mem2);
}
